// S5SSM_21809843929342
// MI455X (gfx1250) — matrix operand order repaired and checked by reading, not compiled, not run on hardware
//
#include <hip/hip_runtime.h>

// ---------------------------------------------------------------------------
// S5 SSM forward (bidirectional) for MI455X / gfx1250.
//   L=16384, H=1024, P=512.  GEMMs in bf16 WMMA (f32 accum), scan in f32.
//   v3: explicit ping-pong double-buffered fragment sets so each set's 12-16
//       B128 loads stay in flight while the other set's WMMAs execute
//       (partial s_wait_loadcnt watermarks instead of wait-0 serialization).
// ---------------------------------------------------------------------------

#define L_DIM 16384
#define H_DIM 1024
#define P_DIM 512
#define T_CHUNK 256
#define NB_CHUNK 64   // L / T
typedef unsigned short u16;
typedef unsigned int   u32;

typedef __attribute__((ext_vector_type(16))) __bf16 v16bf;
typedef __attribute__((ext_vector_type(8)))  float  v8f;

union FragBF {
  v16bf v;
  uint4 q[2];
};

__device__ __forceinline__ u16 f2bf(float f) {
  u32 u = __float_as_uint(f);
  u32 r = (u + 0x7FFFu + ((u >> 16) & 1u)) >> 16;   // round-to-nearest-even
  return (u16)r;
}
__device__ __forceinline__ float bf2f(u16 b) {
  return __uint_as_float(((u32)b) << 16);
}

// A fragment: 16x32 bf16, row-major source (lda elems). lane<16: rows 0..15,
// K = {0..7, 16..23}; lane>=16: same rows, K = {8..15, 24..31}.
__device__ __forceinline__ void load_a_frag(FragBF& f, const u16* __restrict__ A,
                                            int lda, int m0, int k0, int lane) {
  const u16* p = A + (size_t)(m0 + (lane & 15)) * lda + k0 + ((lane >> 4) << 3);
  f.q[0] = *reinterpret_cast<const uint4*>(p);       // K = kb .. kb+7
  f.q[1] = *reinterpret_cast<const uint4*>(p + 16);  // K = kb+16 .. kb+23
}

// B fragment: 32x16 bf16. Source stored N-major: Bop[n*ldb + k].
// lane<16: col n=lane, K = {0..7, 16..23}; lane>=16: col n=lane-16, K = {8..15, 24..31}.
__device__ __forceinline__ void load_b_frag(FragBF& f, const u16* __restrict__ B,
                                            int ldb, int n0, int k0, int lane) {
  const u16* p = B + (size_t)(n0 + (lane & 15)) * ldb + k0 + ((lane >> 4) << 3);
  f.q[0] = *reinterpret_cast<const uint4*>(p);       // K = kb .. kb+7
  f.q[1] = *reinterpret_cast<const uint4*>(p + 16);  // K = kb+16 .. kb+23
}

#define WMMA_BF16(A, B, C)                                                     \
  __builtin_amdgcn_wmma_f32_16x16x32_bf16(false, (A).v, false, (B).v,          \
                                          (short)0, (C), false, false)

// ---------------------------------------------------------------------------
// 1) Discretization: Lambda_bar, f = (Lambda_bar-1)/Lambda, Lambda_bar^T
// ---------------------------------------------------------------------------
__global__ void s5_discretize(const float* __restrict__ lam_re,
                              const float* __restrict__ lam_im,
                              const float* __restrict__ log_step,
                              float* __restrict__ LbRe, float* __restrict__ LbIm,
                              float* __restrict__ fRe,  float* __restrict__ fIm,
                              float* __restrict__ ApRe, float* __restrict__ ApIm) {
  int p = blockIdx.x * blockDim.x + threadIdx.x;
  if (p >= P_DIM) return;
  float step = __expf(log_step[p]);          // STEP_RESCALE == 1.0
  float lr = lam_re[p], li = lam_im[p];
  float er = __expf(lr * step);
  float lbr = er * __cosf(li * step);
  float lbi = er * __sinf(li * step);
  LbRe[p] = lbr; LbIm[p] = lbi;
  // f = (Lb - 1) / (lr + i*li)
  float nr = lbr - 1.0f, ni = lbi;
  float den = 1.0f / (lr * lr + li * li);
  fRe[p] = (nr * lr + ni * li) * den;
  fIm[p] = (ni * lr - nr * li) * den;
  // Lambda_bar^T via 8 complex squarings (T = 256 = 2^8), matches scan numerics
  float ar = lbr, ai = lbi;
  #pragma unroll
  for (int i = 0; i < 8; ++i) {
    float tr = ar * ar - ai * ai;
    float ti = 2.0f * ar * ai;
    ar = tr; ai = ti;
  }
  ApRe[p] = ar; ApIm[p] = ai;
}

// ---------------------------------------------------------------------------
// 2) B_bar -> bf16 operand planes, N-major (P x H): Bop[p*H + h]
// ---------------------------------------------------------------------------
__global__ void s5_bbar(const float* __restrict__ B,
                        const float* __restrict__ fRe, const float* __restrict__ fIm,
                        u16* __restrict__ BopRe, u16* __restrict__ BopIm) {
  int p = blockIdx.x;
  float fr = fRe[p], fi = fIm[p];
  for (int h = threadIdx.x; h < H_DIM; h += blockDim.x) {
    size_t idx = ((size_t)p * H_DIM + h);
    float br = B[idx * 2], bi = B[idx * 2 + 1];
    BopRe[idx] = f2bf(fr * br - fi * bi);
    BopIm[idx] = f2bf(fr * bi + fi * br);
  }
}

// ---------------------------------------------------------------------------
// 3) C operand planes (H x 2P), N-major; imag plane pre-NEGATED so GEMM2
//    chains into one accumulator (bf16 WMMA has no operand-negate).
// ---------------------------------------------------------------------------
__global__ void s5_cprep(const float* __restrict__ C1, const float* __restrict__ C2,
                         u16* __restrict__ CopRe, u16* __restrict__ CopImN) {
  int idx = blockIdx.x * blockDim.x + threadIdx.x;   // over H*2P = 1M
  if (idx >= H_DIM * 2 * P_DIM) return;
  int h = idx >> 10;         // /1024
  int k = idx & 1023;
  const float* src = (k < P_DIM) ? &C1[((size_t)h * P_DIM + k) * 2]
                                 : &C2[((size_t)h * P_DIM + (k - P_DIM)) * 2];
  CopRe[idx]  = f2bf(src[0]);
  CopImN[idx] = f2bf(-src[1]);
}

// ---------------------------------------------------------------------------
// 4) u f32 -> bf16 (vectorized)
// ---------------------------------------------------------------------------
__global__ void s5_uconv(const float* __restrict__ U, u16* __restrict__ U16, int n4) {
  int i = blockIdx.x * blockDim.x + threadIdx.x;
  if (i >= n4) return;
  float4 v = reinterpret_cast<const float4*>(U)[i];
  uint2 o;
  o.x = (u32)f2bf(v.x) | ((u32)f2bf(v.y) << 16);
  o.y = (u32)f2bf(v.z) | ((u32)f2bf(v.w) << 16);
  reinterpret_cast<uint2*>(U16)[i] = o;
}

// ---------------------------------------------------------------------------
// 5) GEMM1: Bu(L x P, complex) = u(L x H) @ B_bar^T.
//    Wave tile M=32 x N=32, two output planes (re/im).
//    Ping-pong fragment sets: while set X's 8 WMMAs run, set Y's 12 B128
//    loads are in flight.  Grid: (L/32, 2) x 256 thr.
// ---------------------------------------------------------------------------
struct Set1 { FragBF a[2], br[2], bi[2]; };

__device__ __forceinline__ void load_set1(Set1& s, const u16* __restrict__ U16,
                                          const u16* __restrict__ BopRe,
                                          const u16* __restrict__ BopIm,
                                          int m0, int n0, int k0, int lane) {
  #pragma unroll
  for (int mt = 0; mt < 2; ++mt)
    load_a_frag(s.a[mt], U16, H_DIM, m0 + mt * 16, k0, lane);
  #pragma unroll
  for (int nt = 0; nt < 2; ++nt) {
    load_b_frag(s.br[nt], BopRe, H_DIM, n0 + nt * 16, k0, lane);
    load_b_frag(s.bi[nt], BopIm, H_DIM, n0 + nt * 16, k0, lane);
  }
}

__device__ __forceinline__ void mma_set1(const Set1& s, v8f (&accRe)[2][2],
                                         v8f (&accIm)[2][2]) {
  #pragma unroll
  for (int mt = 0; mt < 2; ++mt)
    #pragma unroll
    for (int nt = 0; nt < 2; ++nt) {
      accRe[mt][nt] = WMMA_BF16(s.a[mt], s.br[nt], accRe[mt][nt]);
      accIm[mt][nt] = WMMA_BF16(s.a[mt], s.bi[nt], accIm[mt][nt]);
    }
}

__global__ void __launch_bounds__(256)
s5_gemm1(const u16* __restrict__ U16, const u16* __restrict__ BopRe,
         const u16* __restrict__ BopIm, float* __restrict__ BuRe,
         float* __restrict__ BuIm) {
  const int wave = threadIdx.x >> 5, lane = threadIdx.x & 31;
  const int m0 = blockIdx.x << 5;                      // 32 rows
  const int n0 = ((blockIdx.y << 3) + wave) << 5;      // 32 cols
  v8f accRe[2][2] = {}, accIm[2][2] = {};              // [mt][nt]

  Set1 s0, s1;
  load_set1(s0, U16, BopRe, BopIm, m0, n0, 0, lane);
  load_set1(s1, U16, BopRe, BopIm, m0, n0, 32, lane);
  // k coverage: loop handles k0, k0+32 for k0 = 0..H-128; tail does H-64, H-32
  for (int k0 = 0; k0 < H_DIM - 96; k0 += 64) {
    mma_set1(s0, accRe, accIm);
    load_set1(s0, U16, BopRe, BopIm, m0, n0, k0 + 64, lane);
    mma_set1(s1, accRe, accIm);
    load_set1(s1, U16, BopRe, BopIm, m0, n0, k0 + 96, lane);
  }
  mma_set1(s0, accRe, accIm);
  mma_set1(s1, accRe, accIm);

  // C/D layout: lane<16 -> rows +0..7, lane>=16 -> rows +8..15; col = lane&15
  #pragma unroll
  for (int mt = 0; mt < 2; ++mt) {
    const int mrow = m0 + mt * 16 + ((lane >> 4) << 3);
    #pragma unroll
    for (int nt = 0; nt < 2; ++nt) {
      const int col = n0 + nt * 16 + (lane & 15);
      #pragma unroll
      for (int r = 0; r < 8; ++r) {
        BuRe[(size_t)(mrow + r) * P_DIM + col] = accRe[mt][nt][r];
        BuIm[(size_t)(mrow + r) * P_DIM + col] = accIm[mt][nt][r];
      }
    }
  }
}

// ---------------------------------------------------------------------------
// 6) Scan pass 1: per-(chunk, direction) local scan, zero carry-in.
// ---------------------------------------------------------------------------
__global__ void s5_scan_local(const float* __restrict__ BuRe, const float* __restrict__ BuIm,
                              const float* __restrict__ LbRe, const float* __restrict__ LbIm,
                              u16* __restrict__ XsRe, u16* __restrict__ XsIm,
                              float* __restrict__ bendRe, float* __restrict__ bendIm) {
  const int p = threadIdx.x;            // 0..511
  const int b = blockIdx.x;             // chunk
  const int d = blockIdx.y;             // 0 = fwd, 1 = bwd
  const float lr = LbRe[p], li = LbIm[p];
  float sr = 0.f, si = 0.f;
  for (int t = 0; t < T_CHUNK; ++t) {
    const int seq = b * T_CHUNK + t;
    const int l = d ? (L_DIM - 1 - seq) : seq;
    const float br = BuRe[(size_t)l * P_DIM + p];
    const float bi = BuIm[(size_t)l * P_DIM + p];
    const float nr = lr * sr - li * si + br;
    const float ni = lr * si + li * sr + bi;
    sr = nr; si = ni;
    const size_t o = (size_t)l * (2 * P_DIM) + d * P_DIM + p;
    XsRe[o] = f2bf(sr);
    XsIm[o] = f2bf(si);
  }
  const size_t idx = ((size_t)d * NB_CHUNK + b) * P_DIM + p;
  bendRe[idx] = sr; bendIm[idx] = si;
}

// ---------------------------------------------------------------------------
// 7) Scan pass 2: sequential combine of NB chunk summaries per channel.
// ---------------------------------------------------------------------------
__global__ void s5_scan_carry(const float* __restrict__ ApRe, const float* __restrict__ ApIm,
                              const float* __restrict__ bendRe, const float* __restrict__ bendIm,
                              float* __restrict__ cinRe, float* __restrict__ cinIm) {
  const int p = threadIdx.x;   // 0..511
  const int d = blockIdx.x;    // 0..1
  const float ar = ApRe[p], ai = ApIm[p];
  float cr = 0.f, ci = 0.f;
  for (int b = 0; b < NB_CHUNK; ++b) {
    const size_t idx = ((size_t)d * NB_CHUNK + b) * P_DIM + p;
    cinRe[idx] = cr; cinIm[idx] = ci;
    const float nr = ar * cr - ai * ci + bendRe[idx];
    const float ni = ar * ci + ai * cr + bendIm[idx];
    cr = nr; ci = ni;
  }
}

// ---------------------------------------------------------------------------
// 8) Scan pass 3: xs[t] += Lambda_bar^(t+1) * carry  (bf16 RMW in place)
// ---------------------------------------------------------------------------
__global__ void s5_scan_apply(const float* __restrict__ LbRe, const float* __restrict__ LbIm,
                              const float* __restrict__ cinRe, const float* __restrict__ cinIm,
                              u16* __restrict__ XsRe, u16* __restrict__ XsIm) {
  const int p = threadIdx.x;
  const int b = blockIdx.x;
  const int d = blockIdx.y;
  const size_t idx = ((size_t)d * NB_CHUNK + b) * P_DIM + p;
  const float cr = cinRe[idx], ci = cinIm[idx];
  if (cr == 0.f && ci == 0.f) return;       // chunk 0: exact zero carry
  const float lr = LbRe[p], li = LbIm[p];
  float pr = lr, pi = li;                    // Lambda_bar^(t+1), t=0
  for (int t = 0; t < T_CHUNK; ++t) {
    const int seq = b * T_CHUNK + t;
    const int l = d ? (L_DIM - 1 - seq) : seq;
    const size_t o = (size_t)l * (2 * P_DIM) + d * P_DIM + p;
    const float xr = bf2f(XsRe[o]) + (pr * cr - pi * ci);
    const float xi = bf2f(XsIm[o]) + (pr * ci + pi * cr);
    XsRe[o] = f2bf(xr);
    XsIm[o] = f2bf(xi);
    const float nr = pr * lr - pi * li;
    const float ni = pr * li + pi * lr;
    pr = nr; pi = ni;
  }
}

// ---------------------------------------------------------------------------
// 9) GEMM2: ys = 2*(Xre @ Cre^T - Xim @ Cim^T) + D.*u
//    Cim pre-negated -> single accumulator chain.  Wave tile M=32 x N=32,
//    ping-pong fragment sets (16 B128 loads in flight per 8 WMMAs).
//    Grid: (L/32, 4) x 256 thr  (4*8 waves*32 = 1024 cols).
// ---------------------------------------------------------------------------
struct Set2 { FragBF ar[2], ai[2], br[2], bi[2]; };

__device__ __forceinline__ void load_set2(Set2& s, const u16* __restrict__ XsRe,
                                          const u16* __restrict__ XsIm,
                                          const u16* __restrict__ CopRe,
                                          const u16* __restrict__ CopImN,
                                          int m0, int n0, int k0, int lane) {
  const int K2P = 2 * P_DIM;
  #pragma unroll
  for (int mt = 0; mt < 2; ++mt) {
    load_a_frag(s.ar[mt], XsRe, K2P, m0 + mt * 16, k0, lane);
    load_a_frag(s.ai[mt], XsIm, K2P, m0 + mt * 16, k0, lane);
  }
  #pragma unroll
  for (int nt = 0; nt < 2; ++nt) {
    load_b_frag(s.br[nt], CopRe,  K2P, n0 + nt * 16, k0, lane);
    load_b_frag(s.bi[nt], CopImN, K2P, n0 + nt * 16, k0, lane);
  }
}

__device__ __forceinline__ void mma_set2(const Set2& s, v8f (&acc)[2][2]) {
  #pragma unroll
  for (int mt = 0; mt < 2; ++mt)
    #pragma unroll
    for (int nt = 0; nt < 2; ++nt) {
      acc[mt][nt] = WMMA_BF16(s.ar[mt], s.br[nt], acc[mt][nt]);
      acc[mt][nt] = WMMA_BF16(s.ai[mt], s.bi[nt], acc[mt][nt]);
    }
}

__global__ void __launch_bounds__(256)
s5_gemm2(const u16* __restrict__ XsRe, const u16* __restrict__ XsIm,
         const u16* __restrict__ CopRe, const u16* __restrict__ CopImN,
         const float* __restrict__ U, const float* __restrict__ D,
         float* __restrict__ Out) {
  const int wave = threadIdx.x >> 5, lane = threadIdx.x & 31;
  const int m0 = blockIdx.x << 5;                      // 32 rows
  const int n0 = ((blockIdx.y << 3) + wave) << 5;      // 32 cols
  const int K2P = 2 * P_DIM;                           // 1024
  v8f acc[2][2] = {};                                  // [mt][nt]

  Set2 s0, s1;
  load_set2(s0, XsRe, XsIm, CopRe, CopImN, m0, n0, 0, lane);
  load_set2(s1, XsRe, XsIm, CopRe, CopImN, m0, n0, 32, lane);
  for (int k0 = 0; k0 < K2P - 96; k0 += 64) {
    mma_set2(s0, acc);
    load_set2(s0, XsRe, XsIm, CopRe, CopImN, m0, n0, k0 + 64, lane);
    mma_set2(s1, acc);
    load_set2(s1, XsRe, XsIm, CopRe, CopImN, m0, n0, k0 + 96, lane);
  }
  mma_set2(s0, acc);
  mma_set2(s1, acc);

  #pragma unroll
  for (int mt = 0; mt < 2; ++mt) {
    const int mrow = m0 + mt * 16 + ((lane >> 4) << 3);
    #pragma unroll
    for (int nt = 0; nt < 2; ++nt) {
      const int col = n0 + nt * 16 + (lane & 15);
      const float dv = D[col];
      #pragma unroll
      for (int r = 0; r < 8; ++r) {
        const size_t o = (size_t)(mrow + r) * H_DIM + col;
        Out[o] = 2.0f * acc[mt][nt][r] + dv * U[o];
      }
    }
  }
}

// ---------------------------------------------------------------------------
// Host-side launch
// ---------------------------------------------------------------------------
extern "C" void kernel_launch(void* const* d_in, const int* in_sizes, int n_in,
                              void* d_out, int out_size, void* d_ws, size_t ws_size,
                              hipStream_t stream) {
  const float* U      = (const float*)d_in[0];   // (L,H)
  const float* LamRe  = (const float*)d_in[1];   // (P)
  const float* LamIm  = (const float*)d_in[2];   // (P)
  const float* Bp     = (const float*)d_in[3];   // (P,H,2)
  const float* C1     = (const float*)d_in[4];   // (H,P,2)
  const float* C2     = (const float*)d_in[5];   // (H,P,2)
  const float* lstep  = (const float*)d_in[6];   // (P,1)
  const float* Dp     = (const float*)d_in[7];   // (H)
  float* Out = (float*)d_out;                    // (L,H)

  // Workspace carve-out (256B aligned)
  char* w = (char*)d_ws;
  size_t o = 0;
  auto carve = [&](size_t bytes) -> void* {
    void* p = (void*)(w + o);
    o += (bytes + 255) & ~(size_t)255;
    return p;
  };
  u16*   U16    = (u16*)  carve((size_t)L_DIM * H_DIM * 2);
  u16*   BopRe  = (u16*)  carve((size_t)P_DIM * H_DIM * 2);
  u16*   BopIm  = (u16*)  carve((size_t)P_DIM * H_DIM * 2);
  u16*   CopRe  = (u16*)  carve((size_t)H_DIM * 2 * P_DIM * 2);
  u16*   CopImN = (u16*)  carve((size_t)H_DIM * 2 * P_DIM * 2);
  float* BuRe   = (float*)carve((size_t)L_DIM * P_DIM * 4);
  float* BuIm   = (float*)carve((size_t)L_DIM * P_DIM * 4);
  u16*   XsRe   = (u16*)  carve((size_t)L_DIM * 2 * P_DIM * 2);
  u16*   XsIm   = (u16*)  carve((size_t)L_DIM * 2 * P_DIM * 2);
  float* LbRe   = (float*)carve(P_DIM * 4);
  float* LbIm   = (float*)carve(P_DIM * 4);
  float* fRe    = (float*)carve(P_DIM * 4);
  float* fIm    = (float*)carve(P_DIM * 4);
  float* ApRe   = (float*)carve(P_DIM * 4);
  float* ApIm   = (float*)carve(P_DIM * 4);
  float* bendRe = (float*)carve((size_t)2 * NB_CHUNK * P_DIM * 4);
  float* bendIm = (float*)carve((size_t)2 * NB_CHUNK * P_DIM * 4);
  float* cinRe  = (float*)carve((size_t)2 * NB_CHUNK * P_DIM * 4);
  float* cinIm  = (float*)carve((size_t)2 * NB_CHUNK * P_DIM * 4);

  // 1) discretize
  s5_discretize<<<2, 256, 0, stream>>>(LamRe, LamIm, lstep, LbRe, LbIm,
                                       fRe, fIm, ApRe, ApIm);
  // 2) B_bar bf16 planes
  s5_bbar<<<P_DIM, 256, 0, stream>>>(Bp, fRe, fIm, BopRe, BopIm);
  // 3) C bf16 planes (imag negated)
  s5_cprep<<<(H_DIM * 2 * P_DIM) / 256, 256, 0, stream>>>(C1, C2, CopRe, CopImN);
  // 4) u -> bf16
  s5_uconv<<<((L_DIM * H_DIM / 4) + 255) / 256, 256, 0, stream>>>(
      U, U16, L_DIM * H_DIM / 4);
  // 5) GEMM1 (bf16 WMMA): M tiles 32 -> 512 blocks; N: 2*8 waves*32 = 512
  s5_gemm1<<<dim3(L_DIM / 32, 2), 256, 0, stream>>>(U16, BopRe, BopIm, BuRe, BuIm);
  // 6-8) bidirectional chunked scan
  s5_scan_local<<<dim3(NB_CHUNK, 2), P_DIM, 0, stream>>>(
      BuRe, BuIm, LbRe, LbIm, XsRe, XsIm, bendRe, bendIm);
  s5_scan_carry<<<2, P_DIM, 0, stream>>>(ApRe, ApIm, bendRe, bendIm, cinRe, cinIm);
  s5_scan_apply<<<dim3(NB_CHUNK, 2), P_DIM, 0, stream>>>(
      LbRe, LbIm, cinRe, cinIm, XsRe, XsIm);
  // 9) GEMM2 (bf16 WMMA): M tiles 32 -> 512 blocks; N: 4*8 waves*32 = 1024
  s5_gemm2<<<dim3(L_DIM / 32, 4), 256, 0, stream>>>(
      XsRe, XsIm, CopRe, CopImN, U, Dp, Out);
}
